// LSTMCellQ_12249246728586
// MI455X (gfx1250) — matrix operand order repaired and checked by reading, not compiled, not run on hardware
//
#include <hip/hip_runtime.h>
#include <cstdint>

// ---------------- problem dims ----------------
#define BSZ 2048
#define HD  1024
#define KD  1024          // IN == H == 1024
// ---------------- GEMM tiling -----------------
#define BM 128            // rows per block
#define BN 32             // per-gate cols per block (x4 gates fused)
#define BK 64             // k per staged slab (2 WMMA k-substeps)
#define AK 72             // padded LDS row stride (bf16 elems, 144B = 9x16B, 36-bank stride)
#define NSTEPS 32         // (2*KD)/BK stages across both phases (ih, hh)

typedef __bf16 v16bf __attribute__((ext_vector_type(16)));
typedef float  v8f   __attribute__((ext_vector_type(8)));

// ---------------- helpers ---------------------
__device__ __forceinline__ unsigned short f32_to_bf16(float f) {
  uint32_t u = __float_as_uint(f);
  uint32_t r = u + 0x7FFFu + ((u >> 16) & 1u);   // round-to-nearest-even
  return (unsigned short)(r >> 16);
}
__device__ __forceinline__ float pact_clamp(float x, float a) {   // pact_a == clamp
  return fminf(fmaxf(x, -a), a);
}
// forward value of quant_pass (STE only changes grads): 8-bit everywhere (IB/WB/ABNM==8)
__device__ __forceinline__ float qpass(float x, float a) {
  const float sd = 128.0f;
  const float lim = 1.0f - 1.0f / sd;
  float xs = fminf(fmaxf(x / a, -lim), lim);
  return rintf(xs * sd) * (a / sd);
}
__device__ __forceinline__ float qw8(float w) {                   // weight/bias quant, range 1.0
  const float sd = 128.0f;
  float c = fminf(fmaxf(w, -1.0f + 1.0f / sd), 1.0f - 1.0f / sd);
  return rintf(c * sd) * (1.0f / sd);
}
__device__ __forceinline__ float sigmoidf_(float x) { return 1.0f / (1.0f + __expf(-x)); }

// CDNA5 async copy: global -> LDS, 16B, tracked by ASYNCcnt (CDNA5 ISA)
__device__ __forceinline__ void async_b128(uint32_t lds_byte, const void* gaddr) {
  asm volatile("global_load_async_to_lds_b128 %0, %1, off"
               :: "v"(lds_byte), "v"(gaddr) : "memory");
}

// ---------------- prep kernels (f32 -> bf16, read-once) ----------------
__global__ void prep_act_kernel(const float* __restrict__ x, const float* __restrict__ h,
                                const float* __restrict__ pa1,
                                unsigned short* __restrict__ xq,
                                unsigned short* __restrict__ hb, int n) {
  int i = blockIdx.x * blockDim.x + threadIdx.x;
  if (i >= n) return;
  float a1 = pa1[0];
  float v  = qpass(pact_clamp(x[i], a1), a1);   // integer-valued -> exact in bf16
  xq[i] = f32_to_bf16(v);
  hb[i] = f32_to_bf16(h[i]);
}

__global__ void prep_w_kernel(const float* __restrict__ wih, const float* __restrict__ whh,
                              unsigned short* __restrict__ qih,
                              unsigned short* __restrict__ qhh, int n) {
  int i = blockIdx.x * blockDim.x + threadIdx.x;
  if (i >= n) return;
  qih[i] = f32_to_bf16(qw8(wih[i]));            // k/128 -> exact in bf16
  qhh[i] = f32_to_bf16(qw8(whh[i]));
}

__global__ void prep_bias_kernel(const float* __restrict__ bih, const float* __restrict__ bhh,
                                 float* __restrict__ bq, int n) {
  int i = blockIdx.x * blockDim.x + threadIdx.x;
  if (i >= n) return;
  bq[i] = qw8(bih[i]) + qw8(bhh[i]);
}

// ---------------- fused WMMA GEMM + LSTM epilogue ----------------
// grid = (HD/BN, BSZ/BM) ; block = 256 threads = 8 waves (wave32)
// Waves partitioned 4(M) x 2(N): each wave owns 32 rows x 16 cols x 4 gates
// -> 2 A-frags + 4 B-frags per 8 WMMAs (33% less LDS fragment traffic than 1x8).
// Block accumulates the SAME 128x32 tile for gates i,j,f,o (cols n0, n0+H, ...)
// so the LSTM pointwise math is fully in-register.
__global__ __launch_bounds__(256)
void lstm_wmma_kernel(const unsigned short* __restrict__ Xq,
                      const unsigned short* __restrict__ Hb,
                      const unsigned short* __restrict__ Wih,
                      const unsigned short* __restrict__ Whh,
                      const float* __restrict__ biasq,
                      const float* __restrict__ cx,
                      float* __restrict__ out_h, float* __restrict__ out_c,
                      const float* pa3, const float* pa4, const float* pa5,
                      const float* pa6, const float* pa7, const float* pa8,
                      const float* pa9, const float* pa10, const float* pa11) {
  __shared__ __align__(16) unsigned short As[2][BM][AK];       // 2 x 128x64 bf16 (padded)
  __shared__ __align__(16) unsigned short Bs[2][4][BN][AK];    // 2 x 4 gates x 32x64 bf16

  const int tid   = threadIdx.x;
  const int wid   = tid >> 5;
  const int lane  = tid & 31;
  const int wid_m = wid >> 1;                  // 0..3 : which 32-row group
  const int wid_n = wid & 1;                   // 0..1 : which 16-col half
  const int n0    = blockIdx.x * BN;
  const int m0    = blockIdx.y * BM;

  // Stage one (A,B) 64-wide k-slab into LDS buffer `buf` with async b128 copies.
  // 8 async instructions per thread => per-wave ASYNCcnt grows by 8 per stage.
  auto stage = [&](int buf, int s) {
    const int phase = s >> 4;                  // 0: x@Wih^T, 1: h@Whh^T
    const int k0    = (s & 15) * BK;
    const unsigned short* Ab = phase ? Hb  : Xq;
    const unsigned short* Wb = phase ? Whh : Wih;
#pragma unroll
    for (int i = 0; i < 4; ++i) {
      int id  = tid + i * 256;                 // 0..1023
      int row = id >> 3;                       // 0..127
      int c   = id & 7;                        // 16B chunk within 128B of k-row
      async_b128((uint32_t)(uintptr_t)&As[buf][row][c * 8],
                 Ab + (size_t)(m0 + row) * KD + k0 + c * 8);
      int g = row >> 5, n = row & 31;
      async_b128((uint32_t)(uintptr_t)&Bs[buf][g][n][c * 8],
                 Wb + (size_t)(g * HD + n0 + n) * KD + k0 + c * 8);
    }
  };

  const v8f vzero = {0.f, 0.f, 0.f, 0.f, 0.f, 0.f, 0.f, 0.f};
  v8f acc[2][4];                               // [M-tile][gate]
#pragma unroll
  for (int t = 0; t < 2; ++t)
#pragma unroll
    for (int g = 0; g < 4; ++g) acc[t][g] = vzero;

  const int ac0 = (lane >> 4) * 8;             // ISA 16-bit A layout: K-chunk base
  const int bkb = (lane >> 4) * 8;             // ISA 16-bit B layout: K-chunk base (matches A)
  const int bn  = wid_n * 16 + (lane & 15);    // B fragment column within Bs tile

  stage(0, 0);
  for (int s = 0; s < NSTEPS; ++s) {
    const int cur = s & 1;
    if (s + 1 < NSTEPS) {
      stage(cur ^ 1, s + 1);
      asm volatile("s_wait_asynccnt 8" ::: "memory");  // current slab's 8 copies done (in-order)
    } else {
      asm volatile("s_wait_asynccnt 0" ::: "memory");
    }
    __syncthreads();

#pragma unroll
    for (int ks = 0; ks < 2; ++ks) {           // two 32-wide WMMA substeps per slab
      const int kb = ks * 32;
      // A fragments 16x32 bf16: lanes 0-15 carry K 0-7/16-23, lanes 16-31 K 8-15/24-31
      v16bf afrag[2];
#pragma unroll
      for (int t = 0; t < 2; ++t) {
        const int arow = wid_m * 32 + t * 16 + (lane & 15);
        uint4* ap = reinterpret_cast<uint4*>(&afrag[t]);
        ap[0] = *reinterpret_cast<const uint4*>(&As[cur][arow][kb + ac0]);
        ap[1] = *reinterpret_cast<const uint4*>(&As[cur][arow][kb + ac0 + 16]);
      }
#pragma unroll
      for (int g = 0; g < 4; ++g) {
        // B fragment 32x16: lane -> N, lane half -> K half, pairs along VGPRs
        v16bf bfrag;
        uint4* bp = reinterpret_cast<uint4*>(&bfrag);
        bp[0] = *reinterpret_cast<const uint4*>(&Bs[cur][g][bn][kb + bkb]);
        bp[1] = *reinterpret_cast<const uint4*>(&Bs[cur][g][bn][kb + bkb + 16]);
#pragma unroll
        for (int t = 0; t < 2; ++t) {
          acc[t][g] = __builtin_amdgcn_wmma_f32_16x16x32_bf16(
              false, afrag[t], false, bfrag, (short)0, acc[t][g], false, false);
        }
      }
    }
    __syncthreads();   // all waves done reading `cur` before it is restaged
  }

  // ---------------- fused LSTM epilogue (pure VALU) ----------------
  const float a3 = *pa3, a4 = *pa4, a5 = *pa5, a6 = *pa6, a7 = *pa7,
              a8 = *pa8, a9 = *pa9, a10 = *pa10, a11 = *pa11;
  const int n   = n0 + wid_n * 16 + (lane & 15);
  const float bi  = biasq[0 * HD + n];
  const float bj  = biasq[1 * HD + n];
  const float bf_ = biasq[2 * HD + n];
  const float bo  = biasq[3 * HD + n];
  const int mwv = m0 + wid_m * 32;
#pragma unroll
  for (int t = 0; t < 2; ++t) {
#pragma unroll
    for (int r = 0; r < 8; ++r) {
      const int m = mwv + t * 16 + r + ((lane >> 4) << 3);  // C layout: lanes 16-31 -> M+8
      const float iv = acc[t][0][r] + bi;
      const float jv = acc[t][1][r] + bj;
      const float fv = acc[t][2][r] + bf_;
      const float ov = acc[t][3][r] + bo;
      const float fg  = qpass(pact_clamp(sigmoidf_(fv), a3), a3);
      const float ig  = qpass(pact_clamp(sigmoidf_(iv), a4), a4);
      const float act = qpass(pact_clamp(tanhf(jv),    a5), a5);
      const float og  = qpass(pact_clamp(sigmoidf_(ov), a6), a6);
      const float cv  = cx[(size_t)m * HD + n];
      const float gated = qpass(pact_clamp(cv * fg,  a7), a7);
      const float actin = qpass(pact_clamp(ig * act, a8), a8);
      const float nc    = qpass(pact_clamp(gated + actin, a9), a9);
      const float actc  = qpass(pact_clamp(tanhf(nc), a10), a10);
      const float nh    = qpass(pact_clamp(actc * og, a11), a11);
      out_h[(size_t)m * HD + n] = nh;
      out_c[(size_t)m * HD + n] = nc;
    }
  }
}

// ---------------- host glue ----------------
extern "C" void kernel_launch(void* const* d_in, const int* in_sizes, int n_in,
                              void* d_out, int out_size, void* d_ws, size_t ws_size,
                              hipStream_t stream) {
  (void)in_sizes; (void)n_in; (void)out_size; (void)ws_size;
  const float* x   = (const float*)d_in[0];
  const float* hx  = (const float*)d_in[1];
  const float* cx  = (const float*)d_in[2];
  const float* wih = (const float*)d_in[3];
  const float* whh = (const float*)d_in[4];
  const float* bih = (const float*)d_in[5];
  const float* bhh = (const float*)d_in[6];
  const float* pa1 = (const float*)d_in[7];
  // a2 (d_in[8]) is unused by the reference body.
  const float* pa3  = (const float*)d_in[9];
  const float* pa4  = (const float*)d_in[10];
  const float* pa5  = (const float*)d_in[11];
  const float* pa6  = (const float*)d_in[12];
  const float* pa7  = (const float*)d_in[13];
  const float* pa8  = (const float*)d_in[14];
  const float* pa9  = (const float*)d_in[15];
  const float* pa10 = (const float*)d_in[16];
  const float* pa11 = (const float*)d_in[17];

  // workspace layout (bf16 working set ~25MB: L2-resident on MI455X's 192MB L2)
  char* ws = (char*)d_ws;
  unsigned short* Xq   = (unsigned short*)ws;                       // 2048x1024 bf16
  unsigned short* Hb   = Xq + (size_t)BSZ * KD;                     // 2048x1024 bf16
  unsigned short* Wihb = Hb + (size_t)BSZ * HD;                     // 4096x1024 bf16
  unsigned short* Whhb = Wihb + (size_t)4 * HD * KD;                // 4096x1024 bf16
  float*          Bq   = (float*)(Whhb + (size_t)4 * HD * KD);      // 4096 f32

  float* out_h = (float*)d_out;
  float* out_c = out_h + (size_t)BSZ * HD;

  const int nact = BSZ * KD;
  prep_act_kernel<<<nact / 256, 256, 0, stream>>>(x, hx, pa1, Xq, Hb, nact);
  const int nw = 4 * HD * KD;
  prep_w_kernel<<<nw / 256, 256, 0, stream>>>(wih, whh, Wihb, Whhb, nw);
  prep_bias_kernel<<<(4 * HD) / 256, 256, 0, stream>>>(bih, bhh, Bq, 4 * HD);

  dim3 grid(HD / BN, BSZ / BM);   // 32 x 16 blocks, 256 threads each
  lstm_wmma_kernel<<<grid, 256, 0, stream>>>(Xq, Hb, Wihb, Whhb, Bq, cx,
                                             out_h, out_c,
                                             pa3, pa4, pa5, pa6, pa7, pa8,
                                             pa9, pa10, pa11);
}
